// RelationalWorldModel_14199161880576
// MI455X (gfx1250) — matrix operand order repaired and checked by reading, not compiled, not run on hardware
//
#include <hip/hip_runtime.h>
#include <hip/hip_bf16.h>
#include <math.h>

#define Hdim 128

typedef __attribute__((ext_vector_type(16))) __bf16 v16bf;
typedef __attribute__((ext_vector_type(8)))  float  v8f;

// Split f32 into bf16 hi (hardware RNE cvt) + bf16 lo (residual):
// x ~= hi + lo captures ~16 mantissa bits -> fp32-class GEMM accuracy from
// 3 bf16 WMMA products (hh + hl + lh). Native casts lower to
// v_cvt_pk_bf16_f32 instead of manual bit-twiddling VALU sequences.
__device__ __forceinline__ void split2(float x, __bf16& hi, __bf16& lo) {
  __bf16 h = (__bf16)x;
  float hf = (float)h;
  hi = h;
  lo = (__bf16)(x - hf);
}

__device__ __forceinline__ v8f wmma_bf16(v16bf a, v16bf b, v8f c) {
  // D(16x16 f32) = A(16x32 bf16) * B(32x16 bf16) + C
  return __builtin_amdgcn_wmma_f32_16x16x32_bf16(false, a, false, b, (short)0, c,
                                                 false, false);
}

// ---------------------------------------------------------------------------
// Prep: build bf16 hi/lo B-fragments (WMMA B layout) for W_src, W_rel, W_self.
// Frag layout: [t(8 n-tiles)][kk(4 K-steps)][lane(32)][elem(16 bf16)]
//   lane: n = lane&15, elems 0..7 = K(kk*32 + (lane>=16 ? 8 : 0) + 0..7), elems 8..15 = K(+16)
//   B[k][n] = W[n][k]  (einsum 'eh,oh->eo': contract over h)
// One frag = 16384 bf16 = 32 KB.
// Global frag map (elements): src_hi@0 src_lo@16384 rel_hi@32768 rel_lo@49152
//                             self_hi@65536 self_lo@81920
// ---------------------------------------------------------------------------
__global__ void prep_weights_kernel(const float* __restrict__ W_msg_w,
                                    const float* __restrict__ W_self_w,
                                    __bf16* __restrict__ frags) {
  int blk = blockIdx.x;      // 0..95
  int mat = blk >> 5;        // 0=W_src, 1=W_rel, 2=W_self
  int tk  = blk & 31;        // t*4 + kk
  int lane = threadIdx.x;    // 0..31
  int nrow  = (tk >> 2) * 16 + (lane & 15);
  int kbase = (tk & 3) * 32 + (lane >> 4) * 8;
  const float* src;
  if (mat == 0)      src = W_msg_w + nrow * 256 + kbase;        // W_msg_w[:, :H]
  else if (mat == 1) src = W_msg_w + nrow * 256 + 128 + kbase;  // W_msg_w[:, H:]
  else               src = W_self_w + nrow * 128 + kbase;
  __bf16* hi = frags + mat * 32768;
  __bf16* lo = hi + 16384;
  int base = (tk * 32 + lane) * 16;
#pragma unroll
  for (int i = 0; i < 16; ++i) {
    __bf16 h, l;
    split2(src[i + (i >= 8 ? 8 : 0)], h, l);
    hi[base + i] = h;
    lo[base + i] = l;
  }
}

__global__ void zero_kernel(float* __restrict__ p, long long n) {
  long long i = (long long)blockIdx.x * blockDim.x + threadIdx.x;
  if (i < n) p[i] = 0.0f;
}

// ---------------------------------------------------------------------------
// Edge pass: one wave per 16-edge tile; fused
//   msg = tanh(src_h @ W_src^T + edge_rel @ W_rel^T + b) * conf
//   agg[dst] += msg   (f32 L2 atomics)
// Weight frags (src_hi/lo, rel_hi/lo = 128 KB) staged once per block in LDS.
// A layout per ISA: lane m=l&15; elems 0..7 = K(kk*32 + hsel*8 + 0..7),
//                   elems 8..15 = K(+16).
// WMMAs interleaved over two accumulators (t, t+1) so consecutive WMMAs are
// independent (no D->C chain stalls / hazard NOPs).
// ---------------------------------------------------------------------------
__global__ __launch_bounds__(256) void edge_msg_kernel(
    const float* __restrict__ h_cur,
    const float* __restrict__ edge_rel,
    const float* __restrict__ edge_conf,
    const float* __restrict__ W_msg_b,
    const long long* __restrict__ edge_src,
    const long long* __restrict__ edge_dst,
    const __bf16* __restrict__ fragsG,  // [src_hi|src_lo|rel_hi|rel_lo]
    float* __restrict__ agg,
    int E_, int tilesE) {
  __shared__ __align__(32) __bf16 smem[4 * 16384];  // 128 KB
  {
    const float4* g4 = (const float4*)fragsG;
    float4* s4 = (float4*)smem;
    for (int i = threadIdx.x; i < 8192; i += 256) s4[i] = g4[i];
  }
  __syncthreads();

  const int wave = threadIdx.x >> 5;
  const int lane = threadIdx.x & 31;
  const int tile = blockIdx.x * 8 + wave;
  if (tile >= tilesE) return;
  const int m    = lane & 15;
  const int hsel = lane >> 4;
  const long long e_base = (long long)tile * 16;
  long long e_row = e_base + m;
  if (e_row >= E_) e_row = E_ - 1;
  const long long srcIdx = edge_src[e_row];
  const float* __restrict__ arow = h_cur + srcIdx * Hdim;
  const float* __restrict__ rrow = edge_rel + e_row * Hdim;

  v8f acc[8] = {};
#pragma unroll
  for (int kk = 0; kk < 4; ++kk) {
    const int kb = kk * 32 + hsel * 8;
    v16bf a_hi, a_lo, r_hi, r_lo;
#pragma unroll
    for (int i = 0; i < 8; ++i) {
      __bf16 h, l;
      split2(arow[kb + i], h, l);      a_hi[i] = h;     a_lo[i] = l;
      split2(arow[kb + 16 + i], h, l); a_hi[8 + i] = h; a_lo[8 + i] = l;
      split2(rrow[kb + i], h, l);      r_hi[i] = h;     r_lo[i] = l;
      split2(rrow[kb + 16 + i], h, l); r_hi[8 + i] = h; r_lo[8 + i] = l;
    }
#pragma unroll
    for (int t = 0; t < 8; t += 2) {
      const int fo0 = ((t * 4 + kk) * 32 + lane) * 16;
      const int fo1 = (((t + 1) * 4 + kk) * 32 + lane) * 16;
      v16bf bsh0 = *(const v16bf*)(smem + fo0);
      v16bf bsh1 = *(const v16bf*)(smem + fo1);
      v16bf bsl0 = *(const v16bf*)(smem + 16384 + fo0);
      v16bf bsl1 = *(const v16bf*)(smem + 16384 + fo1);
      v16bf brh0 = *(const v16bf*)(smem + 32768 + fo0);
      v16bf brh1 = *(const v16bf*)(smem + 32768 + fo1);
      v16bf brl0 = *(const v16bf*)(smem + 49152 + fo0);
      v16bf brl1 = *(const v16bf*)(smem + 49152 + fo1);
      acc[t]     = wmma_bf16(a_hi, bsh0, acc[t]);
      acc[t + 1] = wmma_bf16(a_hi, bsh1, acc[t + 1]);
      acc[t]     = wmma_bf16(a_lo, bsh0, acc[t]);
      acc[t + 1] = wmma_bf16(a_lo, bsh1, acc[t + 1]);
      acc[t]     = wmma_bf16(a_hi, bsl0, acc[t]);
      acc[t + 1] = wmma_bf16(a_hi, bsl1, acc[t + 1]);
      acc[t]     = wmma_bf16(r_hi, brh0, acc[t]);
      acc[t + 1] = wmma_bf16(r_hi, brh1, acc[t + 1]);
      acc[t]     = wmma_bf16(r_lo, brh0, acc[t]);
      acc[t + 1] = wmma_bf16(r_lo, brh1, acc[t + 1]);
      acc[t]     = wmma_bf16(r_hi, brl0, acc[t]);
      acc[t + 1] = wmma_bf16(r_hi, brl1, acc[t + 1]);
    }
  }

  // C layout: lane l, reg j -> (edge row = j + hsel*8, out col = t*16 + (l&15))
  float conf[8];
  long long dsts[8];
  bool valid[8];
#pragma unroll
  for (int j = 0; j < 8; ++j) {
    long long ej = e_base + hsel * 8 + j;
    valid[j] = (ej < E_);
    long long ec = valid[j] ? ej : (long long)(E_ - 1);
    conf[j] = edge_conf[ec];
    dsts[j] = edge_dst[ec];
  }
#pragma unroll
  for (int t = 0; t < 8; ++t) {
    const int o = t * 16 + m;
    const float bias = W_msg_b[o];
#pragma unroll
    for (int j = 0; j < 8; ++j) {
      if (valid[j]) {
        float v = tanhf(acc[t][j] + bias) * conf[j];
        unsafeAtomicAdd(agg + dsts[j] * Hdim + o, v);
      }
    }
  }
}

// ---------------------------------------------------------------------------
// Node pass: h_out = tanh(h @ W_self^T + b + agg); also zeroes agg for next
// step (each element touched by exactly one lane).
// ---------------------------------------------------------------------------
__global__ __launch_bounds__(256) void node_update_kernel(
    const float* __restrict__ h_cur,
    const float* __restrict__ W_self_b,
    const __bf16* __restrict__ fragsG,  // [self_hi|self_lo]
    float* __restrict__ agg,
    float* __restrict__ h_out,
    int N_, int tilesN) {
  __shared__ __align__(32) __bf16 smem[2 * 16384];  // 64 KB
  {
    const float4* g4 = (const float4*)fragsG;
    float4* s4 = (float4*)smem;
    for (int i = threadIdx.x; i < 4096; i += 256) s4[i] = g4[i];
  }
  __syncthreads();

  const int wave = threadIdx.x >> 5;
  const int lane = threadIdx.x & 31;
  const int tile = blockIdx.x * 8 + wave;
  if (tile >= tilesN) return;
  const int m    = lane & 15;
  const int hsel = lane >> 4;
  const long long n_base = (long long)tile * 16;
  long long n_row = n_base + m;
  if (n_row >= N_) n_row = N_ - 1;
  const float* __restrict__ arow = h_cur + n_row * Hdim;

  v8f acc[8] = {};
#pragma unroll
  for (int kk = 0; kk < 4; ++kk) {
    const int kb = kk * 32 + hsel * 8;
    v16bf a_hi, a_lo;
#pragma unroll
    for (int i = 0; i < 8; ++i) {
      __bf16 h, l;
      split2(arow[kb + i], h, l);      a_hi[i] = h;     a_lo[i] = l;
      split2(arow[kb + 16 + i], h, l); a_hi[8 + i] = h; a_lo[8 + i] = l;
    }
#pragma unroll
    for (int t = 0; t < 8; t += 2) {
      const int fo0 = ((t * 4 + kk) * 32 + lane) * 16;
      const int fo1 = (((t + 1) * 4 + kk) * 32 + lane) * 16;
      v16bf bh0 = *(const v16bf*)(smem + fo0);
      v16bf bh1 = *(const v16bf*)(smem + fo1);
      v16bf bl0 = *(const v16bf*)(smem + 16384 + fo0);
      v16bf bl1 = *(const v16bf*)(smem + 16384 + fo1);
      acc[t]     = wmma_bf16(a_hi, bh0, acc[t]);
      acc[t + 1] = wmma_bf16(a_hi, bh1, acc[t + 1]);
      acc[t]     = wmma_bf16(a_lo, bh0, acc[t]);
      acc[t + 1] = wmma_bf16(a_lo, bh1, acc[t + 1]);
      acc[t]     = wmma_bf16(a_hi, bl0, acc[t]);
      acc[t + 1] = wmma_bf16(a_hi, bl1, acc[t + 1]);
    }
  }

#pragma unroll
  for (int t = 0; t < 8; ++t) {
    const int o = t * 16 + m;
    const float bias = W_self_b[o];
#pragma unroll
    for (int j = 0; j < 8; ++j) {
      long long row = n_base + hsel * 8 + j;
      if (row < N_) {
        long long addr = row * Hdim + o;
        float v = tanhf(acc[t][j] + bias + agg[addr]);
        h_out[addr] = v;
        agg[addr] = 0.0f;  // ready for next step
      }
    }
  }
}

// ---------------------------------------------------------------------------
extern "C" void kernel_launch(void* const* d_in, const int* in_sizes, int n_in,
                              void* d_out, int out_size, void* d_ws, size_t ws_size,
                              hipStream_t stream) {
  const float*     node_states = (const float*)d_in[0];
  const float*     edge_rel    = (const float*)d_in[1];
  const float*     edge_conf   = (const float*)d_in[2];
  const float*     W_msg_w     = (const float*)d_in[3];
  const float*     W_msg_b     = (const float*)d_in[4];
  const float*     W_self_w    = (const float*)d_in[5];
  const float*     W_self_b    = (const float*)d_in[6];
  const long long* edge_src    = (const long long*)d_in[7];
  const long long* edge_dst    = (const long long*)d_in[8];
  float* h_final = (float*)d_out;

  const int E_ = in_sizes[7];
  const int N_ = in_sizes[0] / Hdim;

  // Workspace map: 6 weight frags (192 KB) | agg (N*H f32) | h1 (N*H f32)
  char* ws = (char*)d_ws;
  __bf16* frags = (__bf16*)ws;
  float*  agg   = (float*)(ws + 196608);
  float*  h1    = (float*)(ws + 196608 + (size_t)N_ * Hdim * sizeof(float));

  prep_weights_kernel<<<96, 32, 0, stream>>>(W_msg_w, W_self_w, frags);

  long long nAgg = (long long)N_ * Hdim;
  zero_kernel<<<(int)((nAgg + 255) / 256), 256, 0, stream>>>(agg, nAgg);

  const int tilesE = (E_ + 15) / 16;
  const int gridE  = (tilesE + 7) / 8;
  const int tilesN = (N_ + 15) / 16;
  const int gridN  = (tilesN + 7) / 8;

  // STEP 1
  edge_msg_kernel<<<gridE, 256, 0, stream>>>(node_states, edge_rel, edge_conf,
                                             W_msg_b, edge_src, edge_dst,
                                             frags, agg, E_, tilesE);
  node_update_kernel<<<gridN, 256, 0, stream>>>(node_states, W_self_b,
                                                frags + 65536, agg, h1, N_, tilesN);
  // STEP 2
  edge_msg_kernel<<<gridE, 256, 0, stream>>>(h1, edge_rel, edge_conf,
                                             W_msg_b, edge_src, edge_dst,
                                             frags, agg, E_, tilesE);
  node_update_kernel<<<gridN, 256, 0, stream>>>(h1, W_self_b,
                                                frags + 65536, agg, h_final,
                                                N_, tilesN);
}
